// OOQKVAttention_18021682775013
// MI455X (gfx1250) — matrix operand order repaired and checked by reading, not compiled, not run on hardware
//
#include <hip/hip_runtime.h>
#include <hip/hip_bf16.h>

// Problem constants (match reference)
#define B_  2
#define S_  2048
#define D_  1024
#define H_  16
#define DH_ 64

typedef __attribute__((ext_vector_type(16))) __bf16          v16bf;
typedef __attribute__((ext_vector_type(8)))  float           v8f;
typedef __attribute__((ext_vector_type(8)))  unsigned short  v8us;

union Frag {
    v16bf bf;
    struct { v8us lo, hi; } h;
    unsigned short us[16];
};

// f32 -> bf16 round-to-nearest-even (scalar, epilogue use)
__device__ __forceinline__ unsigned short f2bf(float f) {
    unsigned int u = __float_as_uint(f);
    u += 0x7FFFu + ((u >> 16) & 1u);
    return (unsigned short)(u >> 16);
}

// pack two f32 -> packed bf16x2 (low = a, high = b); lets the compiler pick
// the packed convert instruction where available
__device__ __forceinline__ unsigned int pkbf(float a, float b) {
    __hip_bfloat162 h2 = __float22bfloat162_rn(make_float2(a, b));
    union { __hip_bfloat162 h; unsigned int u; } c;
    c.h = h2;
    return c.u;
}

__device__ __forceinline__ v8f wmma_bf16(const Frag& a, const Frag& b, v8f c) {
    // 8 args: (neg_a, A, neg_b, B, c_mod, C, reuse_a, reuse_b)
    return __builtin_amdgcn_wmma_f32_16x16x32_bf16(
        false, a.bf, false, b.bf, (short)0, c, false, false);
}

// ---------------------------------------------------------------------------
// Staging helper for kernel 1: one 64x32 x-tile (row-major, bf16) and one
// 32x64 W-tile stored k-major ([n][k], bf16) into LDS.
//   x: thread t -> row = t>>2, kseg = (t&3)*8 : two b128 loads, one b128 store
//   W: thread t -> kpair p = t>>4, n = (t&15)*4 : two b128 loads (rows k=2p,
//      2p+1), four b32 stores (bf16 pair (k=2p, 2p+1) at columns n..n+3)
// ---------------------------------------------------------------------------
__device__ __forceinline__ void stage_tiles(
    unsigned short* lx, unsigned short* lw,
    const float* __restrict__ Xp, const float* __restrict__ Wp, int tid)
{
    const float4 x0 = *(const float4*)(Xp);
    const float4 x1 = *(const float4*)(Xp + 4);
    union { v8us v; unsigned int u[4]; } pk;
    pk.u[0] = pkbf(x0.x, x0.y);
    pk.u[1] = pkbf(x0.z, x0.w);
    pk.u[2] = pkbf(x1.x, x1.y);
    pk.u[3] = pkbf(x1.z, x1.w);
    const int r = tid >> 2, kseg = (tid & 3) * 8;
    *(v8us*)&lx[r * 32 + kseg] = pk.v;

    const float4 w0 = *(const float4*)(Wp);        // k = 2p
    const float4 w1 = *(const float4*)(Wp + D_);   // k = 2p+1
    const int p = tid >> 4, n = (tid & 15) * 4;
    unsigned int* lw32 = (unsigned int*)lw;        // [n][k-pair] view
    lw32[(n + 0) * 16 + p] = pkbf(w0.x, w1.x);
    lw32[(n + 1) * 16 + p] = pkbf(w0.y, w1.y);
    lw32[(n + 2) * 16 + p] = pkbf(w0.z, w1.z);
    lw32[(n + 3) * 16 + p] = pkbf(w0.w, w1.w);
}

// ---------------------------------------------------------------------------
// Kernel 1: fused QKV projection.  C[64x64] tile per block, 8 waves (4x2),
// each wave produces 16x32 via two bf16 WMMAs per K-step of 32.
// Double-buffered LDS: stage tile i+1 while computing tile i.
// z = 0: Q (+bias, exact GELU)  -> Q  [B,H,S,64]  bf16
// z = 1: K (+bias)              -> K  [B,H,S,64]  bf16
// z = 2: V (+bias)              -> Vt [B,H,64,S]  bf16  (transposed for PV)
// ---------------------------------------------------------------------------
__global__ __launch_bounds__(256) void qkv_gemm(
    const float* __restrict__ X,
    const float* __restrict__ Wq, const float* __restrict__ bq,
    const float* __restrict__ Wk, const float* __restrict__ bk,
    const float* __restrict__ Wv, const float* __restrict__ bv,
    unsigned short* __restrict__ Qout,
    unsigned short* __restrict__ Kout,
    unsigned short* __restrict__ Vout)
{
    __shared__ __align__(16) unsigned short lx[2][64 * 32];
    __shared__ __align__(16) unsigned short lw[2][64 * 32];

    const int tid  = threadIdx.x;
    const int wave = tid >> 5;
    const int lane = tid & 31;
    const int wm   = wave >> 1;          // 0..3  (16-row subtile)
    const int wn   = wave & 1;           // 0..1  (32-col subtile)
    const int half = lane >> 4;
    const int m    = lane & 15;

    const int row0  = blockIdx.x * 64;   // token tile
    const int n0    = blockIdx.y * 64;   // output-feature tile
    const int which = blockIdx.z;        // 0=Q 1=K 2=V

    const float* W    = (which == 0) ? Wq : (which == 1) ? Wk : Wv;
    const float* bias = (which == 0) ? bq : (which == 1) ? bk : bv;

    // loop-invariant per-thread staging base pointers (k0 = 0)
    const float* Xp = X + (size_t)(row0 + (tid >> 2)) * D_ + (tid & 3) * 8;
    const float* Wp = W + (size_t)(2 * (tid >> 4)) * D_ + n0 + (tid & 15) * 4;

    v8f acc0 = {}; v8f acc1 = {};

    stage_tiles(lx[0], lw[0], Xp, Wp, tid);

    int buf = 0;
    for (int k0 = 0; k0 < D_; k0 += 32, buf ^= 1) {
        __syncthreads();   // staging of buf done; reads of buf^1 done
        if (k0 + 32 < D_)
            stage_tiles(lx[buf ^ 1], lw[buf ^ 1],
                        Xp + (k0 + 32), Wp + (size_t)(k0 + 32) * D_, tid);

        const unsigned short* lxc = lx[buf];
        const unsigned short* lwc = lw[buf];

        // A fragment (16x32): lane m / m+16 hold row m, K 0..15 / 16..31
        Frag a;
        {
            const unsigned short* ap = &lxc[(wm * 16 + m) * 32 + half * 16];
            a.h.lo = *(const v8us*)(ap);
            a.h.hi = *(const v8us*)(ap + 8);
        }
        // B fragments (32x16): lane n holds col n, K 0..15 (half 0) / 16..31 (half 1)
        Frag b0, b1;
        {
            const unsigned short* bp0 = &lwc[(wn * 32 + m) * 32 + half * 16];
            b0.h.lo = *(const v8us*)(bp0);
            b0.h.hi = *(const v8us*)(bp0 + 8);
            const unsigned short* bp1 = &lwc[(wn * 32 + 16 + m) * 32 + half * 16];
            b1.h.lo = *(const v8us*)(bp1);
            b1.h.hi = *(const v8us*)(bp1 + 8);
        }
        acc0 = wmma_bf16(a, b0, acc0);
        acc1 = wmma_bf16(a, b1, acc1);
    }

    // Epilogue: bias (+GELU for Q), convert to bf16, scatter to head-major layouts.
#pragma unroll
    for (int sub = 0; sub < 2; ++sub) {
        v8f acc = sub ? acc1 : acc0;
        const int col = n0 + wn * 32 + sub * 16 + m;   // 0..1023
        const float bcol = bias[col];
        const int h  = col >> 6;
        const int dh = col & 63;
#pragma unroll
        for (int r = 0; r < 8; ++r) {
            const int t  = row0 + wm * 16 + r + 8 * half; // global token
            const int b  = t >> 11;                       // /S_
            const int s  = t & (S_ - 1);
            const int bh = b * H_ + h;
            float v = acc[r] + bcol;
            if (which == 0)  // exact GELU: 0.5*x*(1+erf(x/sqrt(2)))
                v = 0.5f * v * (1.0f + erff(v * 0.70710678118654752f));
            const unsigned short uv = f2bf(v);
            if (which == 2)
                Vout[((size_t)bh * DH_ + dh) * S_ + s] = uv;       // Vt [B,H,64,S]
            else if (which == 1)
                Kout[((size_t)bh * S_ + s) * DH_ + dh] = uv;       // K  [B,H,S,64]
            else
                Qout[((size_t)bh * S_ + s) * DH_ + dh] = uv;       // Q  [B,H,S,64]
        }
    }
}

// ---------------------------------------------------------------------------
// Kernel 2: flash attention (no 1/sqrt(d) scale, per reference).
// One wave = 16 queries of one (b,h); keys streamed in chunks of 32.
// scores: 4 WMMAs (DH=64 split as 2 K-steps, 2 key groups)
// PV:     4 WMMAs (4 dh tiles of 16), P round-trips per-wave LDS C->A layout.
// ---------------------------------------------------------------------------
__global__ __launch_bounds__(256) void attn(
    const unsigned short* __restrict__ Qb,   // [B*H, S, 64]
    const unsigned short* __restrict__ Kb,   // [B*H, S, 64]
    const unsigned short* __restrict__ Vtb,  // [B*H, 64, S]
    float* __restrict__ out)                 // [B, S, H, 64]
{
    __shared__ __align__(16) unsigned short lp[8][16 * 32];  // per-wave P tile

    const int tid  = threadIdx.x;
    const int wave = tid >> 5;
    const int lane = tid & 31;
    const int half = lane >> 4;
    const int m    = lane & 15;

    const int bh = blockIdx.x;            // 0..31
    const int b  = bh >> 4;
    const int h  = bh & (H_ - 1);
    const int q0 = blockIdx.y * 128 + wave * 16;

    const unsigned short* Qp = Qb  + ((size_t)bh * S_ + q0) * DH_;
    const unsigned short* Kp = Kb  + (size_t)bh * S_ * DH_;
    const unsigned short* Vp = Vtb + (size_t)bh * DH_ * S_;

    // Q fragments for dh 0..31 and 32..63 (A layout, loaded once)
    Frag aQ0, aQ1;
    {
        const unsigned short* p = Qp + m * DH_;
        aQ0.h.lo = *(const v8us*)(p + half * 16);
        aQ0.h.hi = *(const v8us*)(p + 8 + half * 16);
        aQ1.h.lo = *(const v8us*)(p + 32 + half * 16);
        aQ1.h.hi = *(const v8us*)(p + 40 + half * 16);
    }

    v8f O[4] = {v8f{}, v8f{}, v8f{}, v8f{}};
    float mrun[8], lrun[8];
#pragma unroll
    for (int r = 0; r < 8; ++r) { mrun[r] = -1e30f; lrun[r] = 0.0f; }

    unsigned short* pl = &lp[wave][0];

    for (int k0 = 0; k0 < S_; k0 += 32) {
        // K fragments: B = K^T (dh x key); lane n holds key k0+g*16+n,
        // contiguous 16 bf16 of K[key, 32*d + 16*half ..]
        Frag bK00, bK01, bK10, bK11;
        {
            const unsigned short* kp0 = Kp + (size_t)(k0 + m) * DH_ + half * 16;
            bK00.h.lo = *(const v8us*)(kp0);       bK00.h.hi = *(const v8us*)(kp0 + 8);
            bK01.h.lo = *(const v8us*)(kp0 + 32);  bK01.h.hi = *(const v8us*)(kp0 + 40);
            const unsigned short* kp1 = Kp + (size_t)(k0 + 16 + m) * DH_ + half * 16;
            bK10.h.lo = *(const v8us*)(kp1);       bK10.h.hi = *(const v8us*)(kp1 + 8);
            bK11.h.lo = *(const v8us*)(kp1 + 32);  bK11.h.hi = *(const v8us*)(kp1 + 40);
        }

        v8f sc0 = {}, sc1 = {};
        sc0 = wmma_bf16(aQ0, bK00, sc0);
        sc0 = wmma_bf16(aQ1, bK01, sc0);
        sc1 = wmma_bf16(aQ0, bK10, sc1);
        sc1 = wmma_bf16(aQ1, bK11, sc1);

        // Online softmax over this 32-key chunk.  Row (r + 8*half) lives across
        // the 16 lanes of a half -> reduce with shfl_xor masks 8..1.
#pragma unroll
        for (int r = 0; r < 8; ++r) {
            float s0 = sc0[r], s1 = sc1[r];
            float t = fmaxf(s0, s1);
#pragma unroll
            for (int msk = 8; msk >= 1; msk >>= 1)
                t = fmaxf(t, __shfl_xor(t, msk, 32));
            const float mnew  = fmaxf(mrun[r], t);
            const float alpha = __expf(mrun[r] - mnew);
            const float p0 = __expf(s0 - mnew);
            const float p1 = __expf(s1 - mnew);
            float rs = p0 + p1;
#pragma unroll
            for (int msk = 8; msk >= 1; msk >>= 1)
                rs += __shfl_xor(rs, msk, 32);
            lrun[r] = lrun[r] * alpha + rs;
            mrun[r] = mnew;
            O[0][r] *= alpha; O[1][r] *= alpha; O[2][r] *= alpha; O[3][r] *= alpha;
            const int row = r + 8 * half;
            pl[row * 32 + m]      = f2bf(p0);   // key group 0
            pl[row * 32 + 16 + m] = f2bf(p1);   // key group 1
        }

        // C-layout -> A-layout round trip through per-wave LDS (same-wave RAW).
        asm volatile("s_wait_dscnt 0" ::: "memory");
        Frag aP;
        {
            const unsigned short* p = pl + m * 32 + half * 16;
            aP.h.lo = *(const v8us*)(p);
            aP.h.hi = *(const v8us*)(p + 8);
        }

        // V fragments: B = V (key x dh); Vt row dh gives contiguous keys.
#pragma unroll
        for (int t2 = 0; t2 < 4; ++t2) {
            const unsigned short* vp = Vp + (size_t)(t2 * 16 + m) * S_ + k0 + half * 16;
            Frag bV;
            bV.h.lo = *(const v8us*)(vp);
            bV.h.hi = *(const v8us*)(vp + 8);
            O[t2] = wmma_bf16(aP, bV, O[t2]);
        }
    }

    // Normalize and write out [B, S, H, 64] fp32.
#pragma unroll
    for (int t2 = 0; t2 < 4; ++t2) {
#pragma unroll
        for (int r = 0; r < 8; ++r) {
            const int s  = q0 + r + 8 * half;
            const int dh = t2 * 16 + m;
            out[(((size_t)b * S_ + s) * H_ + h) * DH_ + dh] = O[t2][r] / lrun[r];
        }
    }
}

// ---------------------------------------------------------------------------
extern "C" void kernel_launch(void* const* d_in, const int* in_sizes, int n_in,
                              void* d_out, int out_size, void* d_ws, size_t ws_size,
                              hipStream_t stream) {
    (void)in_sizes; (void)n_in; (void)out_size; (void)ws_size;
    const float* x  = (const float*)d_in[0];
    const float* Wq = (const float*)d_in[1];
    const float* bq = (const float*)d_in[2];
    const float* Wk = (const float*)d_in[3];
    const float* bk = (const float*)d_in[4];
    const float* Wv = (const float*)d_in[5];
    const float* bv = (const float*)d_in[6];

    const size_t elems = (size_t)B_ * S_ * D_;        // 4M bf16 elements each
    unsigned short* Qw = (unsigned short*)d_ws;
    unsigned short* Kw = Qw + elems;
    unsigned short* Vw = Kw + elems;

    dim3 g1(B_ * S_ / 64, D_ / 64, 3);                // 64 x 16 x 3
    qkv_gemm<<<g1, 256, 0, stream>>>(x, Wq, bq, Wk, bk, Wv, bv, Qw, Kw, Vw);

    dim3 g2(B_ * H_, S_ / 128);                       // 32 x 16
    attn<<<g2, 256, 0, stream>>>(Qw, Kw, Vw, (float*)d_out);
}
